// DecoderBlock_89343909691605
// MI455X (gfx1250) — matrix operand order repaired and checked by reading, not compiled, not run on hardware
//
#include <hip/hip_runtime.h>

#define DM   1024
#define SSEQ 2048
#define BBAT 2
#define TT   4096   // BBAT*SSEQ
#define HH   16
#define DK   64
#define EE   8
#define HID  64

typedef unsigned short u16t;
typedef __attribute__((ext_vector_type(16))) __bf16 v16bf;
typedef __attribute__((ext_vector_type(8)))  float  v8f;

union FragU { v16bf v; uint4 q[2]; };

__device__ inline u16t f2bf(float f) {
  unsigned u = __float_as_uint(f);
  unsigned r = u + 0x7fffu + ((u >> 16) & 1u);
  return (u16t)(r >> 16);
}

// A-matrix fragment 16x32 bf16: lane = row (lane&15); lanes 0-15 hold
// K={0..7,16..23}, lanes 16-31 hold K={8..15,24..31} (ISA A layout).
__device__ inline v16bf load_a_frag(const u16t* base, int ld) {
  int lane = threadIdx.x & 31;
  const u16t* p = base + (size_t)(lane & 15) * ld + ((lane >> 4) << 3);
  FragU f;
  f.q[0] = *(const uint4*)(p);
  f.q[1] = *(const uint4*)(p + 16);
  return f.v;
}
// B-matrix fragment 32x16 bf16: lane = column; lanes 0-15 hold
// K={0..7,16..23}, lanes 16-31 hold K={8..15,24..31}. Source "Bt" = N x K.
__device__ inline v16bf load_b_frag(const u16t* base, int ld) {
  int lane = threadIdx.x & 31;
  const u16t* p = base + (size_t)(lane & 15) * ld + ((lane >> 4) << 3);
  FragU f;
  f.q[0] = *(const uint4*)(p);
  f.q[1] = *(const uint4*)(p + 16);
  return f.v;
}
__device__ inline v8f wmma_bf16(v16bf a, v16bf b, v8f c) {
  return __builtin_amdgcn_wmma_f32_16x16x32_bf16(false, a, false, b, (short)0, c,
                                                 false, false);
}

// gfx1250 async global->LDS copy (16B per lane), tracked by ASYNCcnt.
__device__ inline void async_b128(void* lds_ptr, const void* gptr) {
  unsigned lds_off = (unsigned)(size_t)lds_ptr;  // low 32 bits = LDS offset
  asm volatile("global_load_async_to_lds_b128 %0, %1, off"
               :: "v"(lds_off), "v"(gptr)
               : "memory");
}
__device__ inline void wait_async0() {
  asm volatile("s_wait_asynccnt 0x0" ::: "memory");
}

// ---------------- elementwise / prep kernels ----------------

__global__ __launch_bounds__(256) void to_bf16_kernel(const float* __restrict__ in,
                                                      u16t* __restrict__ out, int n) {
  int i = blockIdx.x * 256 + threadIdx.x;
  if (i < n) out[i] = f2bf(in[i]);
}

// out[n*K + k] = bf16(in[k*N + n]) : produce N x K ("Bt") from K x N f32
__global__ __launch_bounds__(256) void transpose_to_bf16_kernel(
    const float* __restrict__ in, u16t* __restrict__ out, int K, int N) {
  int idx = blockIdx.x * 256 + threadIdx.x;
  if (idx >= K * N) return;
  int n = idx / K, k = idx % K;
  out[idx] = f2bf(in[(size_t)k * N + n]);
}

// Vt[(b*DM + d)*SSEQ + s] = V[(b*SSEQ + s)*DM + d]  (bf16 -> bf16)
__global__ __launch_bounds__(256) void vtrans_kernel(const u16t* __restrict__ v,
                                                     u16t* __restrict__ vt) {
  int idx = blockIdx.x * 256 + threadIdx.x;  // over BBAT*DM*SSEQ
  int s = idx % SSEQ;
  int rest = idx / SSEQ;
  int d = rest % DM;
  int b = rest / DM;
  vt[idx] = v[((size_t)b * SSEQ + s) * DM + d];
}

// ---------------- LDS-staged double-buffered bf16 WMMA GEMM ----------------
// C(MxN) = A(MxK) * B, with B given transposed (Bt: N x K, bf16).
// Block tile 128x128, 8 waves (4M x 2N), wave tile 32x64, K-step 32.
// Tiles staged via global_load_async_to_lds_b128 (ASYNCcnt) with 2 buffers.
template <bool HASBIAS, bool RELU, bool SCALE, bool ACC, bool WF32, bool WB16>
__global__ __launch_bounds__(256) void gemm_bf16_kernel(
    const u16t* __restrict__ A, const u16t* __restrict__ Bt,
    const float* __restrict__ bias, const float* __restrict__ rowscale,
    float* __restrict__ Cf, u16t* __restrict__ Cb, int M, int N, int K) {
  __shared__ __align__(16) u16t sA[2][128 * 32];
  __shared__ __align__(16) u16t sB[2][128 * 32];
  int tid = threadIdx.x;
  int lane = tid & 31, w = tid >> 5;
  int m0b = blockIdx.y * 128;
  int n0b = blockIdx.x * 128;
  int mw = (w & 3) * 32;   // wave tile origin inside block tile
  int nw = (w >> 2) * 64;
  bool active = (n0b + nw) < N;

  // staging work: 512 16B chunks per tile, 2 A-chunks + 2 B-chunks per thread
  int ar0 = tid >> 2, ac0 = (tid & 3) * 8;
  int ar1 = ar0 + 64, ac1 = ac0;
  int bg0 = n0b + ar0; if (bg0 >= N) bg0 = N - 1;   // clamp (N=64 GEMMs)
  int bg1 = n0b + ar1; if (bg1 >= N) bg1 = N - 1;

  int nk = K / 32;
  // prologue: stage k=0 into buffer 0
  {
    const int k = 0;
    async_b128(&sA[0][ar0 * 32 + ac0], A + (size_t)(m0b + ar0) * K + k + ac0);
    async_b128(&sA[0][ar1 * 32 + ac1], A + (size_t)(m0b + ar1) * K + k + ac1);
    async_b128(&sB[0][ar0 * 32 + ac0], Bt + (size_t)bg0 * K + k + ac0);
    async_b128(&sB[0][ar1 * 32 + ac1], Bt + (size_t)bg1 * K + k + ac1);
  }

  v8f acc[2][4] = {};
  for (int it = 0; it < nk; ++it) {
    wait_async0();       // my chunks of buffer it&1 have landed
    __syncthreads();     // everyone's chunks landed; prev reads of it+1 done
    if (it + 1 < nk) {
      int buf = (it + 1) & 1, k = (it + 1) * 32;
      async_b128(&sA[buf][ar0 * 32 + ac0], A + (size_t)(m0b + ar0) * K + k + ac0);
      async_b128(&sA[buf][ar1 * 32 + ac1], A + (size_t)(m0b + ar1) * K + k + ac1);
      async_b128(&sB[buf][ar0 * 32 + ac0], Bt + (size_t)bg0 * K + k + ac0);
      async_b128(&sB[buf][ar1 * 32 + ac1], Bt + (size_t)bg1 * K + k + ac1);
    }
    if (active) {
      int buf = it & 1;
      v16bf af[2], bfr[4];
#pragma unroll
      for (int i = 0; i < 2; ++i)
        af[i] = load_a_frag(&sA[buf][(mw + i * 16) * 32], 32);
#pragma unroll
      for (int j = 0; j < 4; ++j)
        bfr[j] = load_b_frag(&sB[buf][(nw + j * 16) * 32], 32);
#pragma unroll
      for (int i = 0; i < 2; ++i)
#pragma unroll
        for (int j = 0; j < 4; ++j) acc[i][j] = wmma_bf16(af[i], bfr[j], acc[i][j]);
    }
  }
  if (!active) return;

  int cn = lane & 15, hs = lane >> 4;
  int m0 = m0b + mw, n0 = n0b + nw;
#pragma unroll
  for (int i = 0; i < 2; ++i)
#pragma unroll
    for (int j = 0; j < 4; ++j) {
      int col = n0 + j * 16 + cn;
      float bv = HASBIAS ? bias[col] : 0.0f;
#pragma unroll
      for (int r = 0; r < 8; ++r) {
        int row = m0 + i * 16 + r + 8 * hs;
        float v = acc[i][j][r] + bv;
        if (RELU) v = fmaxf(v, 0.0f);
        if (SCALE) v *= rowscale[row];
        size_t o = (size_t)row * N + col;
        if (WF32) Cf[o] = ACC ? (Cf[o] + v) : v;
        if (WB16) Cb[o] = f2bf(v);
      }
    }
}

// ---------------- flash attention (one wave = 16-row Q tile) ----------------
__global__ __launch_bounds__(256) void attn_kernel(
    const u16t* __restrict__ Qb, const u16t* __restrict__ Kb,
    const u16t* __restrict__ Vt, u16t* __restrict__ ctx) {
  __shared__ __align__(16) u16t Pl[8 * 512];  // per-wave 16x32 bf16 P tile
  int lane = threadIdx.x & 31, w = threadIdx.x >> 5;
  int wt = blockIdx.x * 8 + w;                 // 4096 wave-tasks
  int qt = wt & 127;
  int h = (wt >> 7) & 15;
  int b = wt >> 11;
  int cn = lane & 15, hs = lane >> 4;
  u16t* P = Pl + w * 512;

  const u16t* qbase = Qb + ((size_t)(b * SSEQ + qt * 16)) * DM + h * 64;
  v16bf a0 = load_a_frag(qbase, DM);
  v16bf a1 = load_a_frag(qbase + 32, DM);

  v8f o[4] = {};
  float mrow[8], lrow[8];
#pragma unroll
  for (int r = 0; r < 8; ++r) { mrow[r] = -1e30f; lrow[r] = 0.0f; }

  int nk = (qt * 16 + 47) >> 5;  // causal: number of 32-key tiles
  for (int kt = 0; kt < nk; ++kt) {
    int key0 = kt * 32;
    v8f sc[2];
#pragma unroll
    for (int sub = 0; sub < 2; ++sub) {
      const u16t* kb = Kb + ((size_t)(b * SSEQ + key0 + sub * 16)) * DM + h * 64;
      v16bf b0 = load_b_frag(kb, DM);
      v16bf b1 = load_b_frag(kb + 32, DM);
      v8f c = {};
      c = wmma_bf16(a0, b0, c);
      c = wmma_bf16(a1, b1, c);
      sc[sub] = c;
    }
#pragma unroll
    for (int sub = 0; sub < 2; ++sub)
#pragma unroll
      for (int r = 0; r < 8; ++r) {
        int qrow = qt * 16 + r + 8 * hs;
        int key = key0 + sub * 16 + cn;
        float s = sc[sub][r] * 0.125f;  // 1/sqrt(64)
        sc[sub][r] = (key <= qrow) ? s : -1e30f;
      }
    float alpha[8];
#pragma unroll
    for (int r = 0; r < 8; ++r) {
      float lm = fmaxf(sc[0][r], sc[1][r]);
#pragma unroll
      for (int d2 = 1; d2 < 16; d2 <<= 1) lm = fmaxf(lm, __shfl_xor(lm, d2, 32));
      float mn = fmaxf(mrow[r], lm);
      alpha[r] = __expf(mrow[r] - mn);
      mrow[r] = mn;
    }
    float rs[8];
#pragma unroll
    for (int r = 0; r < 8; ++r) rs[r] = 0.0f;
#pragma unroll
    for (int sub = 0; sub < 2; ++sub)
#pragma unroll
      for (int r = 0; r < 8; ++r) {
        float p = __expf(sc[sub][r] - mrow[r]);
        rs[r] += p;
        P[(r + 8 * hs) * 32 + sub * 16 + cn] = f2bf(p);
      }
#pragma unroll
    for (int r = 0; r < 8; ++r) {
      float t = rs[r];
#pragma unroll
      for (int d2 = 1; d2 < 16; d2 <<= 1) t += __shfl_xor(t, d2, 32);
      lrow[r] = lrow[r] * alpha[r] + t;
    }
#pragma unroll
    for (int j = 0; j < 4; ++j)
#pragma unroll
      for (int r = 0; r < 8; ++r) o[j][r] *= alpha[r];

    asm volatile("s_wait_dscnt 0" ::: "memory");  // same-wave DS ordering
    v16bf ap = load_a_frag(P, 32);
#pragma unroll
    for (int j = 0; j < 4; ++j) {
      const u16t* vb = Vt + ((size_t)b * DM + h * 64 + j * 16) * SSEQ + key0;
      v16bf bv = load_b_frag(vb, SSEQ);
      o[j] = wmma_bf16(ap, bv, o[j]);
    }
  }

  float inv[8];
#pragma unroll
  for (int r = 0; r < 8; ++r) inv[r] = 1.0f / lrow[r];
#pragma unroll
  for (int j = 0; j < 4; ++j)
#pragma unroll
    for (int r = 0; r < 8; ++r) {
      int qrow = qt * 16 + r + 8 * hs;
      ctx[((size_t)(b * SSEQ + qrow)) * DM + h * 64 + j * 16 + cn] =
          f2bf(o[j][r] * inv[r]);
    }
}

// ---------------- layernorm (one block per token) ----------------
__global__ __launch_bounds__(256) void ln_kernel(
    const float* __restrict__ a, const float* __restrict__ b,
    const float* __restrict__ gamma, const float* __restrict__ beta,
    float* __restrict__ outf, u16t* __restrict__ outb) {
  __shared__ float red[16];
  __shared__ float stats[2];
  int t = blockIdx.x, tid = threadIdx.x;
  int lane = tid & 31, w = tid >> 5;
  float s1 = 0.0f, s2 = 0.0f;
  for (int d = tid; d < DM; d += 256) {
    float v = a[(size_t)t * DM + d] + b[(size_t)t * DM + d];
    s1 += v; s2 += v * v;
  }
#pragma unroll
  for (int d2 = 1; d2 < 32; d2 <<= 1) { s1 += __shfl_xor(s1, d2, 32); s2 += __shfl_xor(s2, d2, 32); }
  if (lane == 0) { red[w] = s1; red[8 + w] = s2; }
  __syncthreads();
  if (tid == 0) {
    float t1 = 0.0f, t2 = 0.0f;
    for (int i = 0; i < 8; ++i) { t1 += red[i]; t2 += red[8 + i]; }
    float mean = t1 / DM;
    float var = t2 / DM - mean * mean;
    stats[0] = mean;
    stats[1] = rsqrtf(var + 1e-5f);
  }
  __syncthreads();
  float mean = stats[0], rstd = stats[1];
  for (int d = tid; d < DM; d += 256) {
    float v = a[(size_t)t * DM + d] + b[(size_t)t * DM + d];
    float y = (v - mean) * rstd * gamma[d] + beta[d];
    outf[(size_t)t * DM + d] = y;
    if (outb) outb[(size_t)t * DM + d] = f2bf(y);
  }
}

// ---------------- MoE gating ----------------
__global__ __launch_bounds__(256) void gate_logits_kernel(
    const float* __restrict__ x1, const float* __restrict__ gw,
    const float* __restrict__ gb, float* __restrict__ logits) {
  int gid = blockIdx.x * 256 + threadIdx.x;  // TT * EE threads
  int t = gid >> 3, e = gid & 7;
  float s = gb[e];
  for (int k = 0; k < DM; ++k) s += x1[(size_t)t * DM + k] * gw[k * EE + e];
  logits[t * EE + e] = s;
}

__global__ __launch_bounds__(256) void gate_top2_kernel(
    const float* __restrict__ logits, float* __restrict__ scales) {
  int t = blockIdx.x * 256 + threadIdx.x;
  if (t >= TT) return;
  float l[EE];
  float m = -1e30f;
#pragma unroll
  for (int e = 0; e < EE; ++e) { l[e] = logits[t * EE + e]; m = fmaxf(m, l[e]); }
  float sum = 0.0f;
#pragma unroll
  for (int e = 0; e < EE; ++e) { l[e] = __expf(l[e] - m); sum += l[e]; }
#pragma unroll
  for (int e = 0; e < EE; ++e) l[e] /= sum;
  int i0 = 0;
#pragma unroll
  for (int e = 1; e < EE; ++e) if (l[e] > l[i0]) i0 = e;
  int i1 = (i0 == 0) ? 1 : 0;
#pragma unroll
  for (int e = 0; e < EE; ++e) if (e != i0 && l[e] > l[i1]) i1 = e;
  float w0 = l[i0], w1 = l[i1];
  float norm = w0 + w1 + 1e-9f;
  w0 /= norm; w1 /= norm;
#pragma unroll
  for (int e = 0; e < EE; ++e)
    scales[(size_t)e * TT + t] = (e == i0) ? w0 : ((e == i1) ? w1 : 0.0f);
}

// moe[t,d] = sum_e scales[e][t] * eb2[e][d]  (bias of selected experts)
__global__ __launch_bounds__(256) void moe_init_kernel(
    const float* __restrict__ scales, const float* __restrict__ eb2,
    float* __restrict__ moe) {
  int idx = blockIdx.x * 256 + threadIdx.x;  // TT*DM
  int d = idx % DM, t = idx / DM;
  float v = 0.0f;
#pragma unroll
  for (int e = 0; e < EE; ++e) v += scales[(size_t)e * TT + t] * eb2[(size_t)e * DM + d];
  moe[idx] = v;
}

// ---------------- host launch ----------------
extern "C" void kernel_launch(void* const* d_in, const int* in_sizes, int n_in,
                              void* d_out, int out_size, void* d_ws, size_t ws_size,
                              hipStream_t stream) {
  const float* x      = (const float*)d_in[0];
  const float* Wq     = (const float*)d_in[2];
  const float* bq     = (const float*)d_in[3];
  const float* Wk     = (const float*)d_in[4];
  const float* bk     = (const float*)d_in[5];
  const float* Wv     = (const float*)d_in[6];
  const float* bv     = (const float*)d_in[7];
  const float* Wo     = (const float*)d_in[8];
  const float* bo     = (const float*)d_in[9];
  const float* gamma1 = (const float*)d_in[10];
  const float* beta1  = (const float*)d_in[11];
  const float* gamma2 = (const float*)d_in[12];
  const float* beta2  = (const float*)d_in[13];
  const float* gate_w = (const float*)d_in[14];
  const float* gate_b = (const float*)d_in[15];
  const float* ew1    = (const float*)d_in[16];
  const float* eb1    = (const float*)d_in[17];
  const float* ew2    = (const float*)d_in[18];
  const float* eb2    = (const float*)d_in[19];
  float* out = (float*)d_out;

  char* wp = (char*)d_ws;
  auto alloc = [&](size_t bytes) -> char* {
    char* p = wp;
    wp += (bytes + 255) & ~(size_t)255;
    return p;
  };
  const size_t TD = (size_t)TT * DM;
  u16t* xb   = (u16t*)alloc(TD * 2);
  u16t* wqT  = (u16t*)alloc((size_t)DM * DM * 2);
  u16t* wkT  = (u16t*)alloc((size_t)DM * DM * 2);
  u16t* wvT  = (u16t*)alloc((size_t)DM * DM * 2);
  u16t* woT  = (u16t*)alloc((size_t)DM * DM * 2);
  u16t* e1T  = (u16t*)alloc((size_t)EE * HID * DM * 2);
  u16t* e2T  = (u16t*)alloc((size_t)EE * DM * HID * 2);
  u16t* qb   = (u16t*)alloc(TD * 2);
  u16t* kb   = (u16t*)alloc(TD * 2);
  u16t* vb   = (u16t*)alloc(TD * 2);
  u16t* vt   = (u16t*)alloc(TD * 2);
  u16t* ctxb = (u16t*)alloc(TD * 2);
  u16t* x1b  = (u16t*)alloc(TD * 2);
  u16t* hbuf = (u16t*)alloc((size_t)EE * TT * HID * 2);
  float* attn   = (float*)alloc(TD * 4);
  float* x1     = (float*)alloc(TD * 4);
  float* moe    = (float*)alloc(TD * 4);
  float* logits = (float*)alloc((size_t)TT * EE * 4);
  float* scales = (float*)alloc((size_t)EE * TT * 4);

  dim3 blk(256);
  to_bf16_kernel<<<TD / 256, blk, 0, stream>>>(x, xb, (int)TD);
  transpose_to_bf16_kernel<<<DM * DM / 256, blk, 0, stream>>>(Wq, wqT, DM, DM);
  transpose_to_bf16_kernel<<<DM * DM / 256, blk, 0, stream>>>(Wk, wkT, DM, DM);
  transpose_to_bf16_kernel<<<DM * DM / 256, blk, 0, stream>>>(Wv, wvT, DM, DM);
  transpose_to_bf16_kernel<<<DM * DM / 256, blk, 0, stream>>>(Wo, woT, DM, DM);
  for (int e = 0; e < EE; ++e) {
    transpose_to_bf16_kernel<<<DM * HID / 256, blk, 0, stream>>>(
        ew1 + (size_t)e * DM * HID, e1T + (size_t)e * HID * DM, DM, HID);
    transpose_to_bf16_kernel<<<HID * DM / 256, blk, 0, stream>>>(
        ew2 + (size_t)e * HID * DM, e2T + (size_t)e * DM * HID, HID, DM);
  }
  // QKV projections: bias + bf16 out
  dim3 gP(DM / 128, TT / 128);
  gemm_bf16_kernel<true, false, false, false, false, true><<<gP, blk, 0, stream>>>(
      xb, wqT, bq, nullptr, nullptr, qb, TT, DM, DM);
  gemm_bf16_kernel<true, false, false, false, false, true><<<gP, blk, 0, stream>>>(
      xb, wkT, bk, nullptr, nullptr, kb, TT, DM, DM);
  gemm_bf16_kernel<true, false, false, false, false, true><<<gP, blk, 0, stream>>>(
      xb, wvT, bv, nullptr, nullptr, vb, TT, DM, DM);
  vtrans_kernel<<<(int)(TD / 256), blk, 0, stream>>>(vb, vt);
  // attention
  attn_kernel<<<BBAT * HH * (SSEQ / 16) / 8, blk, 0, stream>>>(qb, kb, vt, ctxb);
  // output projection (bias, f32 out) + residual LN1
  gemm_bf16_kernel<true, false, false, false, true, false><<<gP, blk, 0, stream>>>(
      ctxb, woT, bo, nullptr, attn, nullptr, TT, DM, DM);
  ln_kernel<<<TT, blk, 0, stream>>>(x, attn, gamma1, beta1, x1, x1b);
  // gating
  gate_logits_kernel<<<TT * EE / 256, blk, 0, stream>>>(x1, gate_w, gate_b, logits);
  gate_top2_kernel<<<(TT + 255) / 256, blk, 0, stream>>>(logits, scales);
  moe_init_kernel<<<(int)(TD / 256), blk, 0, stream>>>(scales, eb2, moe);
  // experts: up-proj (bias+relu+gate-scale, bf16 out) then down-proj (f32 acc)
  dim3 gE1(1, TT / 128);
  dim3 gE2(DM / 128, TT / 128);
  for (int e = 0; e < EE; ++e) {
    gemm_bf16_kernel<true, true, true, false, false, true><<<gE1, blk, 0, stream>>>(
        x1b, e1T + (size_t)e * HID * DM, eb1 + (size_t)e * HID,
        scales + (size_t)e * TT, nullptr, hbuf + (size_t)e * TT * HID, TT, HID, DM);
  }
  for (int e = 0; e < EE; ++e) {
    gemm_bf16_kernel<false, false, false, true, true, false><<<gE2, blk, 0, stream>>>(
        hbuf + (size_t)e * TT * HID, e2T + (size_t)e * DM * HID, nullptr,
        nullptr, moe, nullptr, TT, DM, HID);
  }
  // residual LN2 -> output
  ln_kernel<<<TT, blk, 0, stream>>>(x1, moe, gamma2, beta2, out, nullptr);
  (void)in_sizes; (void)n_in; (void)out_size; (void)ws_size;
}
